// DirectionalPropagation_72971494359293
// MI455X (gfx1250) — compile-verified, matrix operand order checked by reading, not run on hardware
//
#include <hip/hip_runtime.h>
#include <hip/hip_bf16.h>

typedef __attribute__((ext_vector_type(16))) _Float16 v16h;
typedef __attribute__((ext_vector_type(8)))  float    v8f;

#define HID 128
#define PRI 8
#define TDIM 64
#define KTOT (2*HID + PRI)   /* 264 */
#define KC1 9                /* ceil(264/32) */
#define KC2 2                /* 64/32 */

// ---------------------------------------------------------------------------
// Kernel 1: fused edge MLP.  Per dom edge:
//   h = relu([x[src],x[dst],attr] @ W1 + b1)         (16x264 * 264x64, WMMA f16)
//   t = sigmoid(h @ W2 + b2)                         (16x64  * 64x64,  WMMA f16)
//   ew_dom[e] = sigmoid(dot(t, Wm) + bm)             (cross-lane reduce)
// One wave per 16-edge tile; 8 waves / block.
// ---------------------------------------------------------------------------
__global__ __launch_bounds__(256) void dp_edge_mlp(
    const float* __restrict__ x,   const float* __restrict__ attr,
    const float* __restrict__ W1,  const float* __restrict__ b1,
    const float* __restrict__ W2,  const float* __restrict__ b2,
    const float* __restrict__ Wm,  const float* __restrict__ bm,
    const int* __restrict__ esrc,  const int* __restrict__ edst,
    float* __restrict__ ew_dom, int E)
{
  // W1/W2 pre-swizzled into B-fragment layout: per (kchunk, ntile, lane)
  // the 16 halves a lane needs are contiguous (32B -> ds_load_b128 x2).
  __shared__ __align__(32) _Float16 sW1[KC1 * 4 * 32 * 16];  // 36 KB
  __shared__ __align__(32) _Float16 sW2[KC2 * 4 * 32 * 16];  //  8 KB
  __shared__ __align__(32) _Float16 sH [8 * 16 * 64];        // 16 KB (per-wave h staging)

  const int tid = threadIdx.x;

  // ---- build swizzled W1 fragments (zero-pad k >= 264) ----
  for (int s = tid; s < KC1 * 4 * 32; s += 256) {
    const int f  = s >> 5, l = s & 31;
    const int kc = f >> 2, nt = f & 3;
    const int nc = l & 15, kh = (l >> 4) * 8;
    _Float16* p = &sW1[s * 16];
#pragma unroll
    for (int j = 0; j < 8; ++j) {
      const int k0 = kc * 32 + kh + j;
      const int k1 = k0 + 16;
      p[j]     = (k0 < KTOT) ? (_Float16)W1[k0 * TDIM + nt * 16 + nc] : (_Float16)0.f;
      p[j + 8] = (k1 < KTOT) ? (_Float16)W1[k1 * TDIM + nt * 16 + nc] : (_Float16)0.f;
    }
  }
  // ---- build swizzled W2 fragments ----
  for (int s = tid; s < KC2 * 4 * 32; s += 256) {
    const int f  = s >> 5, l = s & 31;
    const int kc = f >> 2, nt = f & 3;
    const int nc = l & 15, kh = (l >> 4) * 8;
    _Float16* p = &sW2[s * 16];
#pragma unroll
    for (int j = 0; j < 8; ++j) {
      const int k0 = kc * 32 + kh + j;
      const int k1 = k0 + 16;
      p[j]     = (_Float16)W2[k0 * TDIM + nt * 16 + nc];
      p[j + 8] = (_Float16)W2[k1 * TDIM + nt * 16 + nc];
    }
  }
  __syncthreads();

  const int lane = tid & 31;
  const int w    = tid >> 5;
  const long long tileBase = (long long)blockIdx.x * 128 + (long long)w * 16;
  if (tileBase >= E) return;          // wave-uniform exit

  const int r  = lane & 15;           // edge row within tile
  const int hi = lane >> 4;           // which K half this lane feeds
  long long e  = tileBase + r;
  if (e >= E) e = E - 1;              // clamp tail (keeps EXEC full; stores guarded)

  const float* rowS = x   + (long long)esrc[e] * HID;
  const float* rowD = x   + (long long)edst[e] * HID;
  const float* rowA = attr + e * PRI;

  // load 8 consecutive feature dims starting at ks (8-aligned -> single region)
  auto load8 = [&](v16h& a, int o, int ks) {
    if (ks < HID) {
      float4 u = *(const float4*)(rowS + ks);
      float4 v = *(const float4*)(rowS + ks + 4);
      a[o+0]=(_Float16)u.x; a[o+1]=(_Float16)u.y; a[o+2]=(_Float16)u.z; a[o+3]=(_Float16)u.w;
      a[o+4]=(_Float16)v.x; a[o+5]=(_Float16)v.y; a[o+6]=(_Float16)v.z; a[o+7]=(_Float16)v.w;
    } else if (ks < 2*HID) {
      float4 u = *(const float4*)(rowD + ks - HID);
      float4 v = *(const float4*)(rowD + ks - HID + 4);
      a[o+0]=(_Float16)u.x; a[o+1]=(_Float16)u.y; a[o+2]=(_Float16)u.z; a[o+3]=(_Float16)u.w;
      a[o+4]=(_Float16)v.x; a[o+5]=(_Float16)v.y; a[o+6]=(_Float16)v.z; a[o+7]=(_Float16)v.w;
    } else if (ks < KTOT) {
      float4 u = *(const float4*)(rowA);
      float4 v = *(const float4*)(rowA + 4);
      a[o+0]=(_Float16)u.x; a[o+1]=(_Float16)u.y; a[o+2]=(_Float16)u.z; a[o+3]=(_Float16)u.w;
      a[o+4]=(_Float16)v.x; a[o+5]=(_Float16)v.y; a[o+6]=(_Float16)v.z; a[o+7]=(_Float16)v.w;
    } else {
#pragma unroll
      for (int j = 0; j < 8; ++j) a[o + j] = (_Float16)0.f;
    }
  };

  // ---- GEMM1: feats(16x264) @ W1(264x64) ----
  v8f c[4] = {};
#pragma unroll
  for (int kc = 0; kc < KC1; ++kc) {
    const int kb = kc * 32 + hi * 8;
    v16h a;
    load8(a, 0, kb);
    load8(a, 8, kb + 16);
#pragma unroll
    for (int nt = 0; nt < 4; ++nt) {
      v16h bfrag = *(const v16h*)&sW1[((kc * 4 + nt) * 32 + lane) * 16];
      c[nt] = __builtin_amdgcn_wmma_f32_16x16x32_f16(
                  false, a, false, bfrag, (short)0, c[nt], false, false);
    }
  }

  // ---- bias + relu, stage h into per-wave LDS (row-major 16x64 halves) ----
  _Float16* sHw = sH + w * (16 * 64);
  const int ncol = lane & 15;
#pragma unroll
  for (int nt = 0; nt < 4; ++nt) {
    const int n = nt * 16 + ncol;
    const float b1v = b1[n];
#pragma unroll
    for (int g = 0; g < 8; ++g) {
      const int m = hi ? (8 + g) : g;
      float hv = c[nt][g] + b1v;
      sHw[m * 64 + n] = (_Float16)fmaxf(hv, 0.f);
    }
  }
  // DS ops are in-order within a wave -> no barrier needed before reload.

  // ---- GEMM2: h(16x64) @ W2(64x64) ----
  v8f d[4] = {};
#pragma unroll
  for (int kc = 0; kc < KC2; ++kc) {
    const int kb = kc * 32 + hi * 8;
    const _Float16* hp = sHw + r * 64;
    v16h a;
#pragma unroll
    for (int j = 0; j < 8; ++j) { a[j] = hp[kb + j]; a[j + 8] = hp[kb + 16 + j]; }
#pragma unroll
    for (int nt = 0; nt < 4; ++nt) {
      v16h bfrag = *(const v16h*)&sW2[((kc * 4 + nt) * 32 + lane) * 16];
      d[nt] = __builtin_amdgcn_wmma_f32_16x16x32_f16(
                  false, a, false, bfrag, (short)0, d[nt], false, false);
    }
  }

  // ---- epilogue: sigmoid, dot with Wm across N=64, sigmoid, store scalar ----
  const float bmv = bm[0];
  float b2v[4], wmv[4];
#pragma unroll
  for (int nt = 0; nt < 4; ++nt) {
    const int n = nt * 16 + ncol;
    b2v[nt] = b2[n];
    wmv[nt] = Wm[n];
  }
#pragma unroll
  for (int g = 0; g < 8; ++g) {
    float p = 0.f;
#pragma unroll
    for (int nt = 0; nt < 4; ++nt) {
      float t = d[nt][g] + b2v[nt];
      t = 1.f / (1.f + __expf(-t));
      p += t * wmv[nt];
    }
    p += __shfl_xor(p, 1, 16);
    p += __shfl_xor(p, 2, 16);
    p += __shfl_xor(p, 4, 16);
    p += __shfl_xor(p, 8, 16);
    if (ncol == 0) {
      const int row = hi ? (8 + g) : g;
      const long long eo = tileBase + row;
      if (eo < E) ew_dom[eo] = 1.f / (1.f + __expf(-(p + bmv)));
    }
  }
}

// ---------------------------------------------------------------------------
// Kernel 2: plain f32 copy (m_next <- m_cur, also folds the maximum(m, agg)).
// ---------------------------------------------------------------------------
__global__ void dp_copy(const float* __restrict__ a, float* __restrict__ b, int n) {
  int i = blockIdx.x * blockDim.x + threadIdx.x;
  if (i < n) b[i] = a[i];
}

// ---------------------------------------------------------------------------
// Kernel 3: one propagation round. msg = m_cur[psrc] * ew_dom[br]; atomic
// max (uint bit-pattern: exact for non-negative f32) into m_next[pdst].
// ---------------------------------------------------------------------------
__global__ void dp_scatter(const float* __restrict__ mcur, float* mnext,
                           const float* __restrict__ ew, const int* __restrict__ br,
                           const int* __restrict__ psrc, const int* __restrict__ pdst,
                           int E)
{
  int e = blockIdx.x * blockDim.x + threadIdx.x;
  if (e >= E) return;
  const float wgt = ew[br[e]];
  const float4* mp = (const float4*)(mcur + (long long)psrc[e] * 8);
  float4 a0 = mp[0], a1 = mp[1];
  unsigned int* o = (unsigned int*)(mnext + (long long)pdst[e] * 8);
  atomicMax(o + 0, __float_as_uint(a0.x * wgt));
  atomicMax(o + 1, __float_as_uint(a0.y * wgt));
  atomicMax(o + 2, __float_as_uint(a0.z * wgt));
  atomicMax(o + 3, __float_as_uint(a0.w * wgt));
  atomicMax(o + 4, __float_as_uint(a1.x * wgt));
  atomicMax(o + 5, __float_as_uint(a1.y * wgt));
  atomicMax(o + 6, __float_as_uint(a1.z * wgt));
  atomicMax(o + 7, __float_as_uint(a1.w * wgt));
}

extern "C" void kernel_launch(void* const* d_in, const int* in_sizes, int n_in,
                              void* d_out, int out_size, void* d_ws, size_t ws_size,
                              hipStream_t stream) {
  const float* x    = (const float*)d_in[0];
  const float* mask = (const float*)d_in[1];
  const float* attr = (const float*)d_in[2];
  const float* W1   = (const float*)d_in[3];
  const float* b1   = (const float*)d_in[4];
  const float* W2   = (const float*)d_in[5];
  const float* b2   = (const float*)d_in[6];
  const float* Wm   = (const float*)d_in[7];
  const float* bm   = (const float*)d_in[8];
  const int* dom    = (const int*)d_in[9];
  const int* br     = (const int*)d_in[10];
  const int* proj   = (const int*)d_in[11];

  const int E = in_sizes[10];         // br_edge_index is (E,)
  const int N = in_sizes[0] / HID;    // x is (N, 128)
  const int NM = N * 8;

  float* ws = (float*)d_ws;
  float* ew = ws;                     // E floats
  float* mA = ws + E;                 // N*8 floats
  float* mB = mA + (size_t)NM;        // N*8 floats
  float* out = (float*)d_out;

  const int* esrc = dom;
  const int* edst = dom + E;
  const int* psrc = proj;
  const int* pdst = proj + E;

  // fused edge MLP -> per-dom-edge scalar edge weight (pre-gather)
  dp_edge_mlp<<<(E + 127) / 128, 256, 0, stream>>>(
      x, attr, W1, b1, W2, b2, Wm, bm, esrc, edst, ew, E);

  const int cb = (NM + 255) / 256;
  const int eb = (E + 255) / 256;

  // m0 = mask
  dp_copy<<<cb, 256, 0, stream>>>(mask, mA, NM);
  // round 1: mB = max(mA, scatter(mA))
  dp_copy<<<cb, 256, 0, stream>>>(mA, mB, NM);
  dp_scatter<<<eb, 256, 0, stream>>>(mA, mB, ew, br, psrc, pdst, E);
  // round 2: mA = max(mB, scatter(mB))
  dp_copy<<<cb, 256, 0, stream>>>(mB, mA, NM);
  dp_scatter<<<eb, 256, 0, stream>>>(mB, mA, ew, br, psrc, pdst, E);
  // round 3 writes straight into d_out; final max(mask, m) is identity
  dp_copy<<<cb, 256, 0, stream>>>(mA, out, NM);
  dp_scatter<<<eb, 256, 0, stream>>>(mA, out, ew, br, psrc, pdst, E);
}
